// TripleGNNFeatureExtractor_19361712570482
// MI455X (gfx1250) — matrix operand order repaired and checked by reading, not compiled, not run on hardware
//
#include <hip/hip_runtime.h>
#include <hip/hip_bf16.h>
#include <math.h>

#define N_NODES 50000
#define N_EDGES 800000
#define IN_DIM  256
#define HID_DIM 256
#define OUT_DIM 128

typedef __attribute__((ext_vector_type(16))) __bf16        v16bf;
typedef __attribute__((ext_vector_type(8)))  float         v8f;
typedef __attribute__((ext_vector_type(4)))  unsigned int  u32x4;

__device__ __forceinline__ unsigned short f2bf(float f) {
    unsigned int u = __float_as_uint(f);
    u += 0x7FFFu + ((u >> 16) & 1u);           // round-to-nearest-even
    return (unsigned short)(u >> 16);
}

// ---------------------------------------------------------------------------
// f32 -> bf16 elementwise convert (optionally with ReLU)
// ---------------------------------------------------------------------------
__global__ void cvt_bf16_kernel(const float* __restrict__ in,
                                unsigned short* __restrict__ out,
                                int n, int do_relu) {
    int i = blockIdx.x * blockDim.x + threadIdx.x;
    if (i >= n) return;
    float v = in[i];
    if (do_relu) v = fmaxf(v, 0.0f);
    out[i] = f2bf(v);
}

// ---------------------------------------------------------------------------
// Pack weight W[K x NC] (row-major f32) into B-fragment-major bf16 layout:
//   Bp[((ktile*nTiles + ntile)*32 + lane)*16 + 2v + {0,1}]
//     lane 0-15 : col n = ntile*16 + lane,      K = ktile*32 + 2v + {0,1}
//     lane 16-31: col n = ntile*16 + lane-16,   K = ktile*32 + 16 + 2v + {0,1}
// so the GEMM loads one contiguous 32B v16bf per lane per K-step.
// ---------------------------------------------------------------------------
__global__ void pack_w_kernel(const float* __restrict__ W,
                              unsigned short* __restrict__ Bp,
                              int K, int NC) {
    int nTiles = NC >> 4;
    int kTiles = K >> 5;
    int total  = kTiles * nTiles * 32;
    int idx = blockIdx.x * blockDim.x + threadIdx.x;
    if (idx >= total) return;
    int lane  = idx & 31;
    int t     = idx >> 5;
    int ntile = t % nTiles;
    int ktile = t / nTiles;
    int n  = ntile * 16 + (lane & 15);
    int kb = ktile * 32 + ((lane >= 16) ? 16 : 0);
    unsigned short* dst = Bp + (size_t)idx * 16;
#pragma unroll
    for (int v = 0; v < 8; ++v) {
        dst[2 * v]     = f2bf(W[(size_t)(kb + 2 * v)     * NC + n]);
        dst[2 * v + 1] = f2bf(W[(size_t)(kb + 2 * v + 1) * NC + n]);
    }
}

// ---------------------------------------------------------------------------
// C[M x NC] = A[M x K](bf16) * Bp(packed bf16)  via v_wmma_f32_16x16x32_bf16
// block = 256 threads = 8 waves; each wave owns one 16x16 output tile.
// grid = (M/16, NC/128)
// ---------------------------------------------------------------------------
__global__ __launch_bounds__(256)
void gemm_wmma_kernel(const unsigned short* __restrict__ A,
                      const unsigned short* __restrict__ Bp,
                      float* __restrict__ C, int K, int NC) {
    const int lane    = threadIdx.x & 31;
    const int wave    = threadIdx.x >> 5;
    const int rowTile = blockIdx.x;
    const int colTile = blockIdx.y * 8 + wave;
    const int nTiles  = NC >> 4;
    const int kTiles  = K >> 5;

    // A fragment addressing (16-bit A 16x32 layout)
    const int m = rowTile * 16 + (lane & 15);
    const unsigned short* Arow = A + (size_t)m * K + ((lane >= 16) ? 16 : 0);
    // packed B: per-lane contiguous 16 bf16 per K-tile
    const unsigned short* Bl = Bp + ((size_t)colTile * 32 + lane) * 16;
    const size_t bStride = (size_t)nTiles * 32 * 16;

    v8f acc = {};
    for (int kt = 0; kt < kTiles; ++kt) {
        union { u32x4 q[2]; v16bf v; } ua;
        ua.q[0] = *(const u32x4*)(Arow + kt * 32);        // K = kb .. kb+7
        ua.q[1] = *(const u32x4*)(Arow + kt * 32 + 8);    // K = kb+8 .. kb+15
        v16bf b = *(const v16bf*)(Bl + (size_t)kt * bStride);
        acc = __builtin_amdgcn_wmma_f32_16x16x32_bf16(
            false, ua.v, false, b, (short)0, acc, false, false);
    }

    // C/D layout: lanes 0-15 rows M=r, lanes 16-31 rows M=8+r; col = lane&15
    float* Cp = C + (size_t)(rowTile * 16 + ((lane >= 16) ? 8 : 0)) * NC
                  + colTile * 16 + (lane & 15);
#pragma unroll
    for (int r = 0; r < 8; ++r)
        Cp[(size_t)r * NC] = acc[r];
}

// ---------------------------------------------------------------------------
// degree accumulation:  deg[dst[e]] += 1
// ---------------------------------------------------------------------------
__global__ void degree_kernel(const int* __restrict__ dst,
                              float* __restrict__ deg, int E) {
    int e = blockIdx.x * blockDim.x + threadIdx.x;
    if (e >= E) return;
    unsafeAtomicAdd(&deg[dst[e]], 1.0f);
}

__global__ void dinv_kernel(const float* __restrict__ deg,
                            float* __restrict__ dinv, int n) {
    int i = blockIdx.x * blockDim.x + threadIdx.x;
    if (i >= n) return;
    dinv[i] = rsqrtf(deg[i] + 1.0f);           // +1 = self loop
}

// ---------------------------------------------------------------------------
// out[i,c] = t[i,c]*dinv[i]^2 + b[c]   (self-loop term + bias, no atomics)
// ---------------------------------------------------------------------------
__global__ void node_init_kernel(const float* __restrict__ t,
                                 const float* __restrict__ dinv,
                                 const float* __restrict__ bias,
                                 float* __restrict__ out, int C, int total) {
    int idx = blockIdx.x * blockDim.x + threadIdx.x;
    if (idx >= total) return;
    int i = idx / C;
    int c = idx - i * C;
    float di = dinv[i];
    out[idx] = t[idx] * di * di + bias[c];
}

// ---------------------------------------------------------------------------
// edge scatter: one wave per edge
//   out[dst,c] += t[src,c] * dinv[src]*dinv[dst]
// ---------------------------------------------------------------------------
__global__ __launch_bounds__(256)
void edge_scatter_kernel(const int* __restrict__ src,
                         const int* __restrict__ dst,
                         const float* __restrict__ t,
                         const float* __restrict__ dinv,
                         float* __restrict__ out, int E, int C) {
    int wave = threadIdx.x >> 5;
    int lane = threadIdx.x & 31;
    int e = blockIdx.x * 8 + wave;
    if (e >= E) return;
    int s = src[e];
    int d = dst[e];
    float nrm = dinv[s] * dinv[d];
    const float* ts = t + (size_t)s * C;
    float* od = out + (size_t)d * C;
    __builtin_prefetch(ts + lane, 0, 1);       // global_prefetch_b8
    for (int c = lane; c < C; c += 32)
        unsafeAtomicAdd(&od[c], ts[c] * nrm);
}

// ---------------------------------------------------------------------------
// attention fusion: one wave per node
// ---------------------------------------------------------------------------
__global__ __launch_bounds__(256)
void attention_kernel(const float* __restrict__ f0,
                      const float* __restrict__ f1,
                      const float* __restrict__ f2,
                      const float* __restrict__ attn_w,
                      const float* __restrict__ attn_b,
                      float* __restrict__ out, int n) {
    int wave = threadIdx.x >> 5;
    int lane = threadIdx.x & 31;
    int i = blockIdx.x * 8 + wave;
    if (i >= n) return;
    const float* F[3] = { f0 + (size_t)i * OUT_DIM,
                          f1 + (size_t)i * OUT_DIM,
                          f2 + (size_t)i * OUT_DIM };
    float s[3];
#pragma unroll
    for (int g = 0; g < 3; ++g) {
        float acc = 0.0f;
        for (int c = lane; c < OUT_DIM; c += 32)
            acc += F[g][c] * attn_w[c];
#pragma unroll
        for (int off = 16; off >= 1; off >>= 1)
            acc += __shfl_xor(acc, off, 32);
        s[g] = acc + attn_b[0];
    }
    float m = fmaxf(s[0], fmaxf(s[1], s[2]));
    float e0 = __expf(s[0] - m), e1 = __expf(s[1] - m), e2 = __expf(s[2] - m);
    float inv = 1.0f / (e0 + e1 + e2);
    float a0 = e0 * inv, a1 = e1 * inv, a2 = e2 * inv;
    float* o = out + (size_t)i * OUT_DIM;
    for (int c = lane; c < OUT_DIM; c += 32)
        o[c] = a0 * F[0][c] + a1 * F[1][c] + a2 * F[2][c];
}

// ---------------------------------------------------------------------------
// host orchestration
// ---------------------------------------------------------------------------
extern "C" void kernel_launch(void* const* d_in, const int* in_sizes, int n_in,
                              void* d_out, int out_size, void* d_ws, size_t ws_size,
                              hipStream_t stream) {
    const float* x        = (const float*)d_in[0];
    const int*   edges[3] = { (const int*)d_in[1], (const int*)d_in[2], (const int*)d_in[3] };
    const float* W1[3]    = { (const float*)d_in[4],  (const float*)d_in[8],  (const float*)d_in[12] };
    const float* B1[3]    = { (const float*)d_in[5],  (const float*)d_in[9],  (const float*)d_in[13] };
    const float* W2[3]    = { (const float*)d_in[6],  (const float*)d_in[10], (const float*)d_in[14] };
    const float* B2[3]    = { (const float*)d_in[7],  (const float*)d_in[11], (const float*)d_in[15] };
    const float* attn_w   = (const float*)d_in[16];
    const float* attn_b   = (const float*)d_in[17];
    float* out = (float*)d_out;

    // workspace carve-up (256B aligned)
    char* base = (char*)d_ws;
    size_t off = 0;
    auto alloc = [&](size_t bytes) -> char* {
        char* p = base + off;
        off = (off + bytes + 255) & ~(size_t)255;
        return p;
    };
    unsigned short* xb   = (unsigned short*)alloc((size_t)N_NODES * IN_DIM * 2);
    unsigned short* hb   = (unsigned short*)alloc((size_t)N_NODES * HID_DIM * 2);
    unsigned short* wb   = (unsigned short*)alloc((size_t)IN_DIM * HID_DIM * 2);
    float* t1            = (float*)alloc((size_t)N_NODES * HID_DIM * 4);   // also reused as t2
    float* h1            = (float*)alloc((size_t)N_NODES * HID_DIM * 4);
    float* feats         = (float*)alloc((size_t)3 * N_NODES * OUT_DIM * 4);
    float* deg           = (float*)alloc((size_t)N_NODES * 4);
    float* dinv          = (float*)alloc((size_t)N_NODES * 4);
    (void)ws_size;

    const int TB = 256;
    const int rowTiles = N_NODES / 16;                         // 3125

    // x -> bf16 once
    {
        int total = N_NODES * IN_DIM;
        cvt_bf16_kernel<<<(total + TB - 1) / TB, TB, 0, stream>>>(x, xb, total, 0);
    }

    for (int g = 0; g < 3; ++g) {
        const int* src = edges[g];
        const int* dst = edges[g] + N_EDGES;
        float* featg = feats + (size_t)g * N_NODES * OUT_DIM;

        // degrees + dinv
        hipMemsetAsync(deg, 0, (size_t)N_NODES * 4, stream);
        degree_kernel<<<(N_EDGES + TB - 1) / TB, TB, 0, stream>>>(dst, deg, N_EDGES);
        dinv_kernel<<<(N_NODES + TB - 1) / TB, TB, 0, stream>>>(deg, dinv, N_NODES);

        // ---- layer 1: t1 = x @ W1 ; h1 = scatter(t1) + b1 ; relu
        {
            int packTotal = (IN_DIM / 32) * (HID_DIM / 16) * 32;
            pack_w_kernel<<<(packTotal + TB - 1) / TB, TB, 0, stream>>>(W1[g], wb, IN_DIM, HID_DIM);
            gemm_wmma_kernel<<<dim3(rowTiles, HID_DIM / 128), TB, 0, stream>>>(xb, wb, t1, IN_DIM, HID_DIM);
            int total = N_NODES * HID_DIM;
            node_init_kernel<<<(total + TB - 1) / TB, TB, 0, stream>>>(t1, dinv, B1[g], h1, HID_DIM, total);
            edge_scatter_kernel<<<(N_EDGES + 7) / 8, TB, 0, stream>>>(src, dst, t1, dinv, h1, N_EDGES, HID_DIM);
            cvt_bf16_kernel<<<(total + TB - 1) / TB, TB, 0, stream>>>(h1, hb, total, 1);  // relu fused
        }

        // ---- layer 2: t2 = relu(h1) @ W2 ; feat = scatter(t2) + b2
        {
            int packTotal = (HID_DIM / 32) * (OUT_DIM / 16) * 32;
            pack_w_kernel<<<(packTotal + TB - 1) / TB, TB, 0, stream>>>(W2[g], wb, HID_DIM, OUT_DIM);
            gemm_wmma_kernel<<<dim3(rowTiles, OUT_DIM / 128), TB, 0, stream>>>(hb, wb, t1, HID_DIM, OUT_DIM);
            int total = N_NODES * OUT_DIM;
            node_init_kernel<<<(total + TB - 1) / TB, TB, 0, stream>>>(t1, dinv, B2[g], featg, OUT_DIM, total);
            edge_scatter_kernel<<<(N_EDGES + 7) / 8, TB, 0, stream>>>(src, dst, t1, dinv, featg, N_EDGES, OUT_DIM);
        }
    }

    // attention fusion
    attention_kernel<<<(N_NODES + 7) / 8, TB, 0, stream>>>(
        feats, feats + (size_t)N_NODES * OUT_DIM, feats + (size_t)2 * N_NODES * OUT_DIM,
        attn_w, attn_b, out, N_NODES);
    (void)in_sizes; (void)n_in; (void)out_size;
}
